// CrossAttention_62405874811588
// MI455X (gfx1250) — matrix operand order repaired and checked by reading, not compiled, not run on hardware
//
#include <hip/hip_runtime.h>

// Point-transformer vector attention, fused single kernel for gfx1250.
// One wave (32 lanes) processes one point (K=32 neighbors, H=32, hidden=128).
// All GEMMs run on v_wmma_f32_16x16x32_f16 (f32 accumulate); the K=3 pos-MLP
// first layer runs in VALU (padding 3->32 would waste 10x matrix throughput).
// Per-wave LDS staging buffers are synchronized with wavefront-scope fences
// only (LDS ops of a wave are in-order; lanes are lockstep) -- no block
// barriers inside the per-point pipeline.

typedef __attribute__((ext_vector_type(16))) _Float16 v16h;
typedef __attribute__((ext_vector_type(8)))  float    v8f;

union AF { v16h v; unsigned u[8]; _Float16 h[16]; };

__device__ __forceinline__ unsigned pk2(float a, float b) {
  union { _Float16 h[2]; unsigned u; } x;
  x.h[0] = (_Float16)a; x.h[1] = (_Float16)b;
  return x.u;
}

__device__ __forceinline__ v8f wmma_f16(const AF& a, const AF& b, v8f c) {
  // (neg_a, A, neg_b, B, c_mod, C, reuse_a, reuse_b)
  return __builtin_amdgcn_wmma_f32_16x16x32_f16(false, a.v, false, b.v,
                                                (short)0, c, false, false);
}

// Wave-local LDS fence: cross-lane visibility within one wave only.
// DS ops of a wave complete in order; this just stops compiler reordering
// and establishes wavefront-scope memory ordering (no s_barrier needed).
__device__ __forceinline__ void wave_lds_fence() {
  __builtin_amdgcn_fence(__ATOMIC_RELEASE, "wavefront");
  __builtin_amdgcn_wave_barrier();
  __builtin_amdgcn_fence(__ATOMIC_ACQUIRE, "wavefront");
}

// B fragment (32(K) x 16(N) f16) from LDS weight stored transposed,
// pair-packed: [N][Ktot/2] u32. B layout matches A: lane half adds K+8;
// regs 4..7 add K+16; pairs contiguous.
__device__ __forceinline__ AF loadB(const unsigned* w, int kpairs, int nbase,
                                    int kbase, int lane) {
  AF b;
  const unsigned* row =
      w + (nbase + (lane & 15)) * kpairs + ((kbase + ((lane >> 4) << 3)) >> 1);
  #pragma unroll
  for (int r = 0; r < 8; ++r) b.u[r] = row[(r & 3) + ((r & 4) << 1)];
  return b;
}

// A fragment (16(M) x 32(K) f16) from LDS row-major [16][ktot] f16 buffer.
// ISA A layout: lane half adds K+8; regs 4..7 add K+16; pairs contiguous.
__device__ __forceinline__ AF loadA_lds(const _Float16* buf, int ktot,
                                        int kbase, int lane) {
  AF a;
  const _Float16* row = buf + (lane & 15) * ktot + kbase;
  const int hi = (lane >> 4) << 3;
  #pragma unroll
  for (int r = 0; r < 8; ++r) {
    int k = ((r & 4) << 2) + hi + ((r & 3) << 1);
    a.u[r] = *(const unsigned*)&row[k];
  }
  return a;
}

// A fragment (16x32) straight from global f32 row-major [rows][32], cvt to f16.
__device__ __forceinline__ AF loadA_gf32(const float* g, int row0, int lane) {
  AF a;
  const float* row = g + (size_t)(row0 + (lane & 15)) * 32;
  const int hi = (lane >> 4) << 3;
  #pragma unroll
  for (int r = 0; r < 8; ++r) {
    int k = ((r & 4) << 2) + hi + ((r & 3) << 1);
    a.h[2 * r]     = (_Float16)row[k];
    a.h[2 * r + 1] = (_Float16)row[k + 1];
  }
  return a;
}

__global__ __launch_bounds__(256) void pt_vec_attn(
    const float* __restrict__ op,  const float* __restrict__ feats,
    const float* __restrict__ rel,
    const float* __restrict__ Wq,  const float* __restrict__ bq,
    const float* __restrict__ Wk,  const float* __restrict__ bk,
    const float* __restrict__ Wv,  const float* __restrict__ bv,
    const float* __restrict__ pW1, const float* __restrict__ pb1,
    const float* __restrict__ pW2, const float* __restrict__ pb2,
    const float* __restrict__ aW1, const float* __restrict__ ab1,
    const float* __restrict__ aW2,
    float* __restrict__ out, int B) {
  // ---- LDS: shared weights (transposed, f16 pair-packed) + per-wave scratch
  __shared__ unsigned s_WkT[32 * 16];    // [N=32][K=32/2]
  __shared__ unsigned s_WvT[32 * 16];
  __shared__ unsigned s_aW1T[128 * 16];  // [N=128][K=32/2]
  __shared__ unsigned s_aW2T[32 * 64];   // [N=32][K=128/2]
  __shared__ unsigned s_pW2T[32 * 64];
  __shared__ float    s_Wq[32 * 32];
  __shared__ float    s_pW1[3 * 128];
  __shared__ float    s_bq[32], s_bk[32], s_bv[32], s_pb2[32];
  __shared__ float    s_pb1[128], s_ab1[128];
  __shared__ _Float16 s_h[8 * 16 * 128]; // per-wave hidden staging [16][128]
  __shared__ _Float16 s_t[8 * 16 * 32];  // per-wave (q-k+pos) staging [16][32]

  const int tid  = threadIdx.x;
  const int lane = tid & 31;
  const int wv   = tid >> 5;

  // ---- cooperative weight staging (once per block)
  for (int i = tid; i < 32 * 32; i += 256) s_Wq[i] = Wq[i];
  for (int i = tid; i < 32 * 16; i += 256) {
    int n = i >> 4, kp = i & 15;
    s_WkT[i] = pk2(Wk[(2 * kp) * 32 + n], Wk[(2 * kp + 1) * 32 + n]);
    s_WvT[i] = pk2(Wv[(2 * kp) * 32 + n], Wv[(2 * kp + 1) * 32 + n]);
  }
  for (int i = tid; i < 128 * 16; i += 256) {
    int n = i >> 4, kp = i & 15;
    s_aW1T[i] = pk2(aW1[(2 * kp) * 128 + n], aW1[(2 * kp + 1) * 128 + n]);
  }
  for (int i = tid; i < 32 * 64; i += 256) {
    int n = i >> 6, kp = i & 63;
    s_aW2T[i] = pk2(aW2[(2 * kp) * 32 + n], aW2[(2 * kp + 1) * 32 + n]);
    s_pW2T[i] = pk2(pW2[(2 * kp) * 32 + n], pW2[(2 * kp + 1) * 32 + n]);
  }
  for (int i = tid; i < 3 * 128; i += 256) s_pW1[i] = pW1[i];
  for (int i = tid; i < 128; i += 256) { s_pb1[i] = pb1[i]; s_ab1[i] = ab1[i]; }
  if (tid < 32) {
    s_bq[tid] = bq[tid]; s_bk[tid] = bk[tid];
    s_bv[tid] = bv[tid]; s_pb2[tid] = pb2[tid];
  }
  __syncthreads();  // only cross-wave barrier: weights become visible

  int p = blockIdx.x * 8 + wv;
  if (p >= B) p = B - 1;  // duplicate work on tail, no divergence

  _Float16* hb = s_h + wv * (16 * 128);
  _Float16* tb = s_t + wv * (16 * 32);

  // ---- query: q[h] = bq[h] + sum_i op[p][i] * Wq[i][h]  (lane h)
  float opv = op[(size_t)p * 32 + lane];
  float qv  = s_bq[lane];
  #pragma unroll
  for (int i = 0; i < 32; ++i)
    qv += __shfl(opv, i, 32) * s_Wq[i * 32 + lane];

  v8f vlog[2][2];  // logits C frags, kept in registers through softmax
  v8f vval[2][2];  // value (+pos+bias) C frags

  #pragma unroll
  for (int mt = 0; mt < 2; ++mt) {   // 16-neighbor M-tiles
    const int k0 = mt * 16;

    // ---- pos MLP layer 1 in VALU (K=3): hidden = relu(rel @ pW1 + pb1)
    {
      int m = lane & 15, ch = lane >> 4;
      const float* rp = rel + ((size_t)p * 32 + k0 + m) * 3;
      float x = rp[0], y = rp[1], z = rp[2];
      #pragma unroll
      for (int j = 0; j < 64; ++j) {
        int c = ch * 64 + j;
        float hv = fmaf(x, s_pW1[c],
                   fmaf(y, s_pW1[128 + c],
                   fmaf(z, s_pW1[256 + c], s_pb1[c])));
        hb[m * 128 + c] = (_Float16)fmaxf(hv, 0.0f);
      }
    }
    wave_lds_fence();

    // ---- pos MLP layer 2: [16,128] @ pW2 -> posC (WMMA x8)
    v8f posC[2] = {{}, {}};
    #pragma unroll
    for (int kc = 0; kc < 4; ++kc) {
      AF ah = loadA_lds(hb, 128, kc * 32, lane);
      #pragma unroll
      for (int nt = 0; nt < 2; ++nt)
        posC[nt] = wmma_f16(ah, loadB(s_pW2T, 64, nt * 16, kc * 32, lane),
                            posC[nt]);
    }

    // ---- key / value GEMMs (WMMA x4), build t = q - key + pos, keep value
    AF af = loadA_gf32(feats + (size_t)p * 32 * 32, k0, lane);
    #pragma unroll
    for (int nt = 0; nt < 2; ++nt) {
      v8f kc_ = {};
      v8f vc_ = {};
      kc_ = wmma_f16(af, loadB(s_WkT, 16, nt * 16, 0, lane), kc_);
      vc_ = wmma_f16(af, loadB(s_WvT, 16, nt * 16, 0, lane), vc_);
      int n = nt * 16 + (lane & 15);
      float qn  = __shfl(qv, n, 32);
      float bkn = s_bk[n], bvn = s_bv[n], pbn = s_pb2[n];
      #pragma unroll
      for (int r = 0; r < 8; ++r) {
        float pos_ = posC[nt][r] + pbn;
        float tv   = qn - (kc_[r] + bkn) + pos_;
        int row = (lane < 16) ? r : r + 8;  // C-frag M mapping
        tb[row * 32 + n] = (_Float16)tv;
        vval[mt][nt][r] = vc_[r] + bvn + pos_;
      }
    }
    wave_lds_fence();

    // ---- attn MLP layer 1: relu(t @ aW1 + ab1) -> hb   (WMMA x8)
    AF at = loadA_lds(tb, 32, 0, lane);
    #pragma unroll
    for (int n8 = 0; n8 < 8; ++n8) {
      v8f hc = {};
      hc = wmma_f16(at, loadB(s_aW1T, 16, n8 * 16, 0, lane), hc);
      int n = n8 * 16 + (lane & 15);
      float b1 = s_ab1[n];
      #pragma unroll
      for (int r = 0; r < 8; ++r) {
        int row = (lane < 16) ? r : r + 8;
        hb[row * 128 + n] = (_Float16)fmaxf(hc[r] + b1, 0.0f);
      }
    }
    wave_lds_fence();

    // ---- attn MLP layer 2: h @ aW2 -> logits (WMMA x8).
    // ab2 is constant per channel across K -> softmax-invariant -> dropped.
    v8f lc[2] = {{}, {}};
    #pragma unroll
    for (int kc = 0; kc < 4; ++kc) {
      AF ah = loadA_lds(hb, 128, kc * 32, lane);
      #pragma unroll
      for (int nt = 0; nt < 2; ++nt)
        lc[nt] = wmma_f16(ah, loadB(s_aW2T, 64, nt * 16, kc * 32, lane),
                          lc[nt]);
    }
    vlog[mt][0] = lc[0];
    vlog[mt][1] = lc[1];
    wave_lds_fence();  // protect hb/tb WAR before next M-tile
  }

  // ---- per-channel softmax over K=32 rows, then sum(attn * value).
  // Lane L and L^16 hold the same column n = nt*16 + (L&15), complementary
  // row halves, so one shfl_xor(16) completes each reduction.
  #pragma unroll
  for (int nt = 0; nt < 2; ++nt) {
    float mx = -3.4e38f;
    #pragma unroll
    for (int mt = 0; mt < 2; ++mt)
      #pragma unroll
      for (int r = 0; r < 8; ++r)
        mx = fmaxf(mx, vlog[mt][nt][r]);
    mx = fmaxf(mx, __shfl_xor(mx, 16, 32));
    float se = 0.0f, sw = 0.0f;
    #pragma unroll
    for (int mt = 0; mt < 2; ++mt)
      #pragma unroll
      for (int r = 0; r < 8; ++r) {
        float e = __expf(vlog[mt][nt][r] - mx);
        se += e;
        sw += e * vval[mt][nt][r];
      }
    se += __shfl_xor(se, 16, 32);
    sw += __shfl_xor(sw, 16, 32);
    if (lane < 16) out[(size_t)p * 32 + nt * 16 + lane] = sw / se;
  }
}

extern "C" void kernel_launch(void* const* d_in, const int* in_sizes, int n_in,
                              void* d_out, int out_size, void* d_ws,
                              size_t ws_size, hipStream_t stream) {
  (void)n_in; (void)out_size; (void)d_ws; (void)ws_size;
  const float* op    = (const float*)d_in[0];
  const float* feats = (const float*)d_in[1];
  const float* rel   = (const float*)d_in[2];
  const float* Wq  = (const float*)d_in[3];
  const float* bq  = (const float*)d_in[4];
  const float* Wk  = (const float*)d_in[5];
  const float* bk  = (const float*)d_in[6];
  const float* Wv  = (const float*)d_in[7];
  const float* bv  = (const float*)d_in[8];
  const float* pW1 = (const float*)d_in[9];
  const float* pb1 = (const float*)d_in[10];
  const float* pW2 = (const float*)d_in[11];
  const float* pb2 = (const float*)d_in[12];
  const float* aW1 = (const float*)d_in[13];
  const float* ab1 = (const float*)d_in[14];
  const float* aW2 = (const float*)d_in[15];
  // d_in[16] (ab2) intentionally unused: per-channel constant, softmax-invariant.
  float* out = (float*)d_out;

  int B = in_sizes[0] / 32;          // op is [B, 32]
  int blocks = (B + 7) / 8;          // 8 waves/block, 1 point per wave
  hipLaunchKernelGGL(pt_vec_attn, dim3(blocks), dim3(256), 0, stream,
                     op, feats, rel, Wq, bq, Wk, bk, Wv, bv,
                     pW1, pb1, pW2, pb2, aW1, ab1, aW2, out, B);
}
